// WorldModel_16449724744327
// MI455X (gfx1250) — matrix operand order repaired and checked by reading, not compiled, not run on hardware
//
#include <hip/hip_runtime.h>
#include <hip/hip_bf16.h>

typedef __bf16 bf16;
typedef __attribute__((ext_vector_type(16))) __bf16 v16bf;
typedef __attribute__((ext_vector_type(8)))  float  v8f;

#define B_  64
#define S_  64
#define HN  1024
#define LC_ 1024
#define G3H 3072
#define NWG 32

// ---------------------------------------------------------------- utilities
static __device__ inline v8f zero_v8f() {
  v8f z;
#pragma unroll
  for (int i = 0; i < 8; ++i) z[i] = 0.f;
  return z;
}

static __device__ inline float wred_max(float v) {
#pragma unroll
  for (int o = 16; o; o >>= 1) v = fmaxf(v, __shfl_xor(v, o, 32));
  return v;
}
static __device__ inline float wred_sum(float v) {
#pragma unroll
  for (int o = 16; o; o >>= 1) v += __shfl_xor(v, o, 32);
  return v;
}

// ------------------------------------------------- WMMA fragment loaders
// A: 16x32 bf16 tile, row-major source A[row][k], row stride lda (elements).
// gfx1250 layout: lanes 0-15 row M=lane hold K {0..7,16..23}; lanes 16-31 row
// M=lane-16 hold K {8..15,24..31}; 2 bf16 per VGPR, K-pairs contiguous.
static __device__ inline v16bf load_afrag(const bf16* A, long lda) {
  const int lane = threadIdx.x & 31;
  const int row  = lane & 15;
  const int hi   = (lane >> 4) & 1;            // 0: K base 0, 1: K base 8
  const unsigned* base = (const unsigned*)(A + (long)row * lda);
  union { unsigned u[8]; v16bf v; } f;
#pragma unroll
  for (int j = 0; j < 4; ++j) f.u[j]     = base[hi * 4 + j];      // K = hi*8 + 2j
#pragma unroll
  for (int j = 0; j < 4; ++j) f.u[4 + j] = base[8 + hi * 4 + j];  // K = 16 + hi*8 + 2j
  return f.v;
}

// B: 32x16 bf16 tile from weights stored row-major W[n][k] (i.e. B^T), so
// each lane (column n = lane&15) reads its K-pairs contiguously.
// lanes 0-15: K {0..7,16..23}; lanes 16-31: K {8..15,24..31}.
static __device__ inline v16bf load_bfrag(const bf16* W, long ldw) {
  const int lane = threadIdx.x & 31;
  const int n    = lane & 15;
  const int kb   = (lane & 16) ? 4 : 0;        // dword offset (8 bf16 -> 4 dwords)
  const unsigned* base = (const unsigned*)(W + (long)n * ldw);
  union { unsigned u[8]; v16bf v; } f;
#pragma unroll
  for (int j = 0; j < 8; ++j) f.u[j] = base[(j >> 2) * 8 + kb + (j & 3)];
  return f.v;
}

// One K=32 chunk of a 32(M) x 64(N) wave tile:
// 2 A-frags (rows m0, m0+16) x 4 B-frags -> 8 WMMAs; each B reused twice.
static __device__ inline void wmma8(const bf16* A0, const bf16* A1, long lda,
                                    const bf16* W, long ldw, v8f acc[8]) {
  v16bf a0 = load_afrag(A0, lda);
  v16bf a1 = load_afrag(A1, lda);
#pragma unroll
  for (int i = 0; i < 4; ++i) {
    v16bf b = load_bfrag(W + (long)(16 * i) * ldw, ldw);
    acc[i] = __builtin_amdgcn_wmma_f32_16x16x32_bf16(
        false, a0, false, b, (short)0, acc[i], false, false);
    acc[4 + i] = __builtin_amdgcn_wmma_f32_16x16x32_bf16(
        false, a1, false, b, (short)0, acc[4 + i], false, false);
  }
}

// ---------------------------------------------------------------- prep
__global__ void f2b_kernel(const float* __restrict__ s, bf16* __restrict__ d, long n) {
  for (long i = (long)blockIdx.x * blockDim.x + threadIdx.x; i < n;
       i += (long)gridDim.x * blockDim.x)
    d[i] = (bf16)s[i];
}

// strided column slice (w_ih[:, :1024] -> bf16 [3072][1024])
__global__ void f2b_cols_kernel(const float* __restrict__ s, int slds,
                                bf16* __restrict__ d, int cols, long n) {
  for (long i = (long)blockIdx.x * blockDim.x + threadIdx.x; i < n;
       i += (long)gridDim.x * blockDim.x) {
    long r = i / cols, c = i % cols;
    d[i] = (bf16)s[r * slds + c];
  }
}

// ---------------------------------------------------------------- LayerNorm
__global__ void __launch_bounds__(256) ln_kernel(const float* __restrict__ obs,
                                                 const float* __restrict__ g,
                                                 const float* __restrict__ bb,
                                                 bf16* __restrict__ xln) {
  const long tok = blockIdx.x;                 // 4096 tokens
  const float* x = obs + tok * HN;
  float s = 0.f, s2 = 0.f;
  for (int i = threadIdx.x; i < HN; i += 256) { float v = x[i]; s += v; s2 += v * v; }
  s = wred_sum(s); s2 = wred_sum(s2);
  __shared__ float ws1[8], ws2[8];
  const int wid = threadIdx.x >> 5;
  if ((threadIdx.x & 31) == 0) { ws1[wid] = s; ws2[wid] = s2; }
  __syncthreads();
  float ts = 0.f, ts2 = 0.f;
#pragma unroll
  for (int i = 0; i < 8; ++i) { ts += ws1[i]; ts2 += ws2[i]; }
  const float mu = ts * (1.f / HN);
  const float var = ts2 * (1.f / HN) - mu * mu;
  const float rstd = rsqrtf(var + 1e-5f);
  for (int i = threadIdx.x; i < HN; i += 256)
    xln[tok * HN + i] = (bf16)((x[i] - mu) * rstd * g[i] + bb[i]);
}

// ---------------------------------------------------------------- generic GEMM
// C[M,N] = act(A[M,K](bf16) @ W[N,K](bf16)^T + bias); one wave per 32x64 tile.
__global__ void __launch_bounds__(256) gemm_bf16_kernel(
    const bf16* __restrict__ A, int lda, const bf16* __restrict__ W, int ldw,
    const float* __restrict__ bias, float* __restrict__ outf,
    bf16* __restrict__ outb, int M, int N, int K, int relu) {
  const int tilesN = N >> 6;
  const int T = (M >> 5) * tilesN;
  const int tile = blockIdx.x * (blockDim.x >> 5) + (threadIdx.x >> 5);
  if (tile >= T) return;
  const int m0 = (tile / tilesN) << 5;
  const int nb = (tile % tilesN) << 6;
  v8f acc[8];
#pragma unroll
  for (int i = 0; i < 8; ++i) acc[i] = zero_v8f();
  for (int k0 = 0; k0 < K; k0 += 32) {
    __builtin_prefetch(A + (long)m0 * lda + k0 + 128, 0, 0);   // next A chunk
    wmma8(A + (long)m0 * lda + k0, A + (long)(m0 + 16) * lda + k0, lda,
          W + (long)nb * ldw + k0, ldw, acc);
  }
  const int lane = threadIdx.x & 31;
  const int ncol = lane & 15;
#pragma unroll
  for (int h = 0; h < 2; ++h) {
    const int mb = m0 + 16 * h + ((lane & 16) ? 8 : 0);
#pragma unroll
    for (int i = 0; i < 4; ++i) {
      const int n = nb + 16 * i + ncol;
      const float bv = bias ? bias[n] : 0.f;
#pragma unroll
      for (int r = 0; r < 8; ++r) {
        float v = acc[4 * h + i][r] + bv;
        if (relu) v = fmaxf(v, 0.f);
        const long o = (long)(mb + r) * N + n;
        if (outf) outf[o] = v;
        if (outb) outb[o] = (bf16)v;
      }
    }
  }
}

// ---------------------------------------------------------------- global barrier
static __device__ inline void gbar(unsigned* cnt, unsigned* gen) {
  __syncthreads();
  if (threadIdx.x == 0) {
    __threadfence();
    unsigned g = __hip_atomic_load(gen, __ATOMIC_ACQUIRE, __HIP_MEMORY_SCOPE_AGENT);
    unsigned a = __hip_atomic_fetch_add(cnt, 1u, __ATOMIC_ACQ_REL, __HIP_MEMORY_SCOPE_AGENT);
    if (a == NWG - 1) {
      __hip_atomic_store(cnt, 0u, __ATOMIC_RELAXED, __HIP_MEMORY_SCOPE_AGENT);
      __hip_atomic_fetch_add(gen, 1u, __ATOMIC_RELEASE, __HIP_MEMORY_SCOPE_AGENT);
    } else {
      while (__hip_atomic_load(gen, __ATOMIC_ACQUIRE, __HIP_MEMORY_SCOPE_AGENT) == g)
        __builtin_amdgcn_s_sleep(1);
    }
    __threadfence();
  }
  __syncthreads();
}

// ---------------------------------------------------------------- recurrent scan
__global__ void __launch_bounds__(256, 1) recurrent_kernel(
    const bf16* __restrict__ encb,      // [B*S, 1024] bf16
    const bf16* __restrict__ postw,     // [1024][2048]
    const bf16* __restrict__ wih,       // [3072][1024]
    const bf16* __restrict__ whh,       // [3072][1024]
    const bf16* __restrict__ priorw,    // [1024][1024]
    const float* __restrict__ post_b, const float* __restrict__ b_ih,
    const float* __restrict__ b_hh, const float* __restrict__ prior_b,
    const float* __restrict__ w_ih_f32, // [3072][1042] fp32
    const int* __restrict__ act_seq,    // [B,S]
    const float* __restrict__ gumbel,   // [B,S,1024]
    const int* __restrict__ tau_p,
    bf16* __restrict__ hcur, float* __restrict__ hfp,
    bf16* __restrict__ decin,           // [B*S][2048]
    float* __restrict__ post_lg, float* __restrict__ prior_lg, // [B*S][1024]
    float* __restrict__ gi, float* __restrict__ gh,            // [64][3072]
    float* __restrict__ out_h, float* __restrict__ out_samp,
    unsigned* __restrict__ sync) {
  unsigned* cnt = sync;
  unsigned* gen = sync + 16;
  const int gwid = blockIdx.x * 8 + (threadIdx.x >> 5);   // 0..255
  const int lane = threadIdx.x & 31;
  const int ncol = lane & 15;
  const float inv_tau = 1.f / (float)tau_p[0];
  const long ldrow = (long)S_ * HN;                       // batch-row stride in enc/logit bufs

  for (int t = 0; t < S_; ++t) {
    // ---- phase 1: posterior logits = [h | enc_t] @ post_w^T + post_b
    for (int job = gwid; job < 32; job += 256) {
      const int m0 = (job >> 4) << 5, nb = (job & 15) << 6;
      v8f acc[8];
#pragma unroll
      for (int i = 0; i < 8; ++i) acc[i] = zero_v8f();
      for (int k0 = 0; k0 < 2048; k0 += 32) {
        const bf16 *A0, *A1; long lda;
        if (k0 < HN) {
          A0 = hcur + (long)m0 * HN + k0;
          A1 = hcur + (long)(m0 + 16) * HN + k0;
          lda = HN;
        } else {
          A0 = encb + ((long)m0 * S_ + t) * HN + (k0 - HN);
          A1 = encb + ((long)(m0 + 16) * S_ + t) * HN + (k0 - HN);
          lda = ldrow;
        }
        wmma8(A0, A1, lda, postw + (long)nb * 2048 + k0, 2048, acc);
      }
#pragma unroll
      for (int h = 0; h < 2; ++h) {
        const int mb = m0 + 16 * h + ((lane & 16) ? 8 : 0);
#pragma unroll
        for (int i = 0; i < 4; ++i) {
          const int n = nb + 16 * i + ncol;
          const float bv = post_b[n];
#pragma unroll
          for (int r = 0; r < 8; ++r)
            post_lg[(long)(mb + r) * ldrow + (long)t * HN + n] = acc[4 * h + i][r] + bv;
        }
      }
    }
    gbar(cnt, gen);

    // ---- phase 2: gumbel-softmax sample over C=32 groups (one wave / group)
    for (int job = gwid; job < B_ * 32; job += 256) {
      const int b = job >> 5, l = job & 31;
      const long idx = ((long)b * S_ + t) * HN + l * 32 + lane;
      const float v = (post_lg[idx] + gumbel[idx]) * inv_tau;
      const float m = wred_max(v);
      const float e = __expf(v - m);
      const float p = e / wred_sum(e);
      out_samp[idx] = p;
      decin[((long)b * S_ + t) * 2048 + HN + l * 32 + lane] = (bf16)p;
    }
    gbar(cnt, gen);

    // ---- phase 3: gi = sample @ Wih^T (+act col +b_ih); gh = h @ Whh^T (+b_hh)
    for (int job = gwid; job < 192; job += 256) {
      const int sel = job / 96, rem = job % 96;
      const int m0 = (rem / 48) << 5, nb = (rem % 48) << 6;
      const bf16 *A0, *A1; long lda; const bf16* W;
      if (sel == 0) {
        A0 = decin + ((long)m0 * S_ + t) * 2048 + HN;
        A1 = decin + ((long)(m0 + 16) * S_ + t) * 2048 + HN;
        lda = (long)S_ * 2048; W = wih;
      } else {
        A0 = hcur + (long)m0 * HN;
        A1 = hcur + (long)(m0 + 16) * HN;
        lda = HN; W = whh;
      }
      v8f acc[8];
#pragma unroll
      for (int i = 0; i < 8; ++i) acc[i] = zero_v8f();
      for (int k0 = 0; k0 < HN; k0 += 32)
        wmma8(A0 + k0, A1 + k0, lda, W + (long)nb * HN + k0, HN, acc);
      float* outp = sel ? gh : gi;
      const float* bias = sel ? b_hh : b_ih;
#pragma unroll
      for (int h = 0; h < 2; ++h) {
        const int mb = m0 + 16 * h + ((lane & 16) ? 8 : 0);
#pragma unroll
        for (int i = 0; i < 4; ++i) {
          const int n = nb + 16 * i + ncol;
          const float bv = bias[n];
#pragma unroll
          for (int r = 0; r < 8; ++r) {
            const int brow = mb + r;
            float v = acc[4 * h + i][r] + bv;
            if (sel == 0) v += w_ih_f32[(long)n * 1042 + HN + act_seq[brow * S_ + t]];
            outp[(long)brow * G3H + n] = v;
          }
        }
      }
    }
    gbar(cnt, gen);

    // ---- phase 4: GRU cell update
    for (int e = blockIdx.x * 256 + threadIdx.x; e < B_ * HN; e += NWG * 256) {
      const int b = e >> 10, j = e & (HN - 1);
      const float ir = gi[(long)b * G3H + j];
      const float iz = gi[(long)b * G3H + HN + j];
      const float in_ = gi[(long)b * G3H + 2 * HN + j];
      const float hr = gh[(long)b * G3H + j];
      const float hz = gh[(long)b * G3H + HN + j];
      const float hn = gh[(long)b * G3H + 2 * HN + j];
      const float r = 1.f / (1.f + __expf(-(ir + hr)));
      const float z = 1.f / (1.f + __expf(-(iz + hz)));
      const float n = tanhf(in_ + r * hn);
      const float hp = hfp[e];
      const float hnew = (1.f - z) * n + z * hp;
      hfp[e] = hnew;
      hcur[e] = (bf16)hnew;
      const long tok = (long)b * S_ + t;
      decin[tok * 2048 + j] = (bf16)hnew;
      out_h[tok * HN + j] = hnew;
    }
    gbar(cnt, gen);

    // ---- phase 5: prior logits = h_new @ prior_w^T + prior_b
    for (int job = gwid; job < 32; job += 256) {
      const int m0 = (job >> 4) << 5, nb = (job & 15) << 6;
      v8f acc[8];
#pragma unroll
      for (int i = 0; i < 8; ++i) acc[i] = zero_v8f();
      for (int k0 = 0; k0 < HN; k0 += 32)
        wmma8(hcur + (long)m0 * HN + k0, hcur + (long)(m0 + 16) * HN + k0, HN,
              priorw + (long)nb * HN + k0, HN, acc);
#pragma unroll
      for (int h = 0; h < 2; ++h) {
        const int mb = m0 + 16 * h + ((lane & 16) ? 8 : 0);
#pragma unroll
        for (int i = 0; i < 4; ++i) {
          const int n = nb + 16 * i + ncol;
          const float bv = prior_b[n];
#pragma unroll
          for (int r = 0; r < 8; ++r)
            prior_lg[(long)(mb + r) * ldrow + (long)t * HN + n] = acc[4 * h + i][r] + bv;
        }
      }
    }
    gbar(cnt, gen);
  }
}

// ---------------------------------------------------------------- KL loss
__global__ void __launch_bounds__(256) kl_kernel(const float* __restrict__ plg,
                                                 const float* __restrict__ qlg,
                                                 float* __restrict__ out_kl) {
  const int tok = blockIdx.x * 8 + (threadIdx.x >> 5);  // 4096 tokens
  const int lane = threadIdx.x & 31;
  float klsum = 0.f;
  for (int l = 0; l < 32; ++l) {
    const long idx = (long)tok * HN + l * 32 + lane;
    const float lp = plg[idx], lq = qlg[idx];
    const float mp = wred_max(lp);
    const float lsp = lp - mp - logf(wred_sum(__expf(lp - mp)));
    const float mq = wred_max(lq);
    const float lsq = lq - mq - logf(wred_sum(__expf(lq - mq)));
    const float klf = wred_sum(__expf(lsp) * (lsp - lsq));
    const float klr = wred_sum(__expf(lsq) * (lsq - lsp));
    klsum += 0.8f * klf + 0.2f * klr;
  }
  if (lane == 0)
    atomicAdd(out_kl, fmaxf(klsum - 1.0f, 0.f) * (1.0f / (B_ * S_)));
}

// ---------------------------------------------------------------- reward head
__global__ void __launch_bounds__(256) reward_kernel(const bf16* __restrict__ decin,
                                                     const float* __restrict__ rw,
                                                     const float* __restrict__ rb,
                                                     float* __restrict__ out) {
  const int tok = blockIdx.x * 8 + (threadIdx.x >> 5);
  const int lane = threadIdx.x & 31;
  float s = 0.f;
  for (int k = lane; k < 2048; k += 32)
    s += (float)decin[(long)tok * 2048 + k] * rw[k];
  s = wred_sum(s);
  if (lane == 0) out[tok] = s + rb[0];
}

// ---------------------------------------------------------------- launch
extern "C" void kernel_launch(void* const* d_in, const int* in_sizes, int n_in,
                              void* d_out, int out_size, void* d_ws, size_t ws_size,
                              hipStream_t stream) {
  const float* obs    = (const float*)d_in[0];
  const int*   actseq = (const int*)d_in[1];
  const float* gumbel = (const float*)d_in[2];
  const int*   tau    = (const int*)d_in[3];
  const float* ln_g   = (const float*)d_in[4];
  const float* ln_b   = (const float*)d_in[5];
  const float* enc_w1 = (const float*)d_in[6];
  const float* enc_b1 = (const float*)d_in[7];
  const float* enc_w2 = (const float*)d_in[8];
  const float* enc_b2 = (const float*)d_in[9];
  const float* w_ih   = (const float*)d_in[10];
  const float* w_hh   = (const float*)d_in[11];
  const float* b_ih   = (const float*)d_in[12];
  const float* b_hh   = (const float*)d_in[13];
  const float* priorw = (const float*)d_in[14];
  const float* priorb = (const float*)d_in[15];
  const float* postw  = (const float*)d_in[16];
  const float* postb  = (const float*)d_in[17];
  const float* decw1  = (const float*)d_in[18];
  const float* decb1  = (const float*)d_in[19];
  const float* decw2  = (const float*)d_in[20];
  const float* decb2  = (const float*)d_in[21];
  const float* reww   = (const float*)d_in[22];
  const float* rewb   = (const float*)d_in[23];

  float* out_recon = (float*)d_out;                      // 4096x1024
  float* out_rew   = out_recon + (long)4096 * 1024;      // 4096
  float* out_kl    = out_rew + 4096;                     // 1
  float* out_h     = out_kl + 1;                         // 4096x1024
  float* out_samp  = out_h + (long)4096 * 1024;          // 4096x1024

  char* w = (char*)d_ws;
  auto carve = [&](size_t bytes) {
    void* p = (void*)w;
    w += (bytes + 255) & ~(size_t)255;
    return p;
  };
  bf16* encw1b  = (bf16*)carve((size_t)1024 * 1024 * 2);
  bf16* encw2b  = (bf16*)carve((size_t)1024 * 1024 * 2);
  bf16* postwb  = (bf16*)carve((size_t)1024 * 2048 * 2);
  bf16* wihb    = (bf16*)carve((size_t)3072 * 1024 * 2);
  bf16* whhb    = (bf16*)carve((size_t)3072 * 1024 * 2);
  bf16* priorwb = (bf16*)carve((size_t)1024 * 1024 * 2);
  bf16* decw1b  = (bf16*)carve((size_t)1024 * 2048 * 2);
  bf16* decw2b  = (bf16*)carve((size_t)1024 * 1024 * 2);
  bf16* xln     = (bf16*)carve((size_t)4096 * 1024 * 2);
  bf16* e1      = (bf16*)carve((size_t)4096 * 1024 * 2);
  bf16* encb    = (bf16*)carve((size_t)4096 * 1024 * 2);
  bf16* decin   = (bf16*)carve((size_t)4096 * 2048 * 2);
  bf16* dbuf    = (bf16*)carve((size_t)4096 * 1024 * 2);
  bf16* hcur    = (bf16*)carve((size_t)64 * 1024 * 2);
  float* hfp    = (float*)carve((size_t)64 * 1024 * 4);
  float* gi     = (float*)carve((size_t)64 * 3072 * 4);
  float* gh     = (float*)carve((size_t)64 * 3072 * 4);
  float* postlg = (float*)carve((size_t)4096 * 1024 * 4);
  float* priolg = (float*)carve((size_t)4096 * 1024 * 4);
  unsigned* sync = (unsigned*)carve(256);

  // zero recurrent state / barrier / kl accumulator (graph-capturable)
  hipMemsetAsync(sync, 0, 256, stream);
  hipMemsetAsync(hcur, 0, (size_t)64 * 1024 * 2, stream);
  hipMemsetAsync(hfp, 0, (size_t)64 * 1024 * 4, stream);
  hipMemsetAsync(out_kl, 0, 4, stream);

  // weight conversion to bf16
  f2b_kernel<<<1024, 256, 0, stream>>>(enc_w1, encw1b, (long)1024 * 1024);
  f2b_kernel<<<1024, 256, 0, stream>>>(enc_w2, encw2b, (long)1024 * 1024);
  f2b_kernel<<<1024, 256, 0, stream>>>(postw, postwb, (long)1024 * 2048);
  f2b_cols_kernel<<<1024, 256, 0, stream>>>(w_ih, 1042, wihb, 1024, (long)3072 * 1024);
  f2b_kernel<<<1024, 256, 0, stream>>>(w_hh, whhb, (long)3072 * 1024);
  f2b_kernel<<<1024, 256, 0, stream>>>(priorw, priorwb, (long)1024 * 1024);
  f2b_kernel<<<1024, 256, 0, stream>>>(decw1, decw1b, (long)1024 * 2048);
  f2b_kernel<<<1024, 256, 0, stream>>>(decw2, decw2b, (long)1024 * 1024);

  // encoder
  ln_kernel<<<4096, 256, 0, stream>>>(obs, ln_g, ln_b, xln);
  gemm_bf16_kernel<<<256, 256, 0, stream>>>(xln, 1024, encw1b, 1024, enc_b1,
                                            nullptr, e1, 4096, 1024, 1024, 1);
  gemm_bf16_kernel<<<256, 256, 0, stream>>>(e1, 1024, encw2b, 1024, enc_b2,
                                            nullptr, encb, 4096, 1024, 1024, 1);

  // recurrent scan (persistent kernel, 32 WGs, global barrier between phases)
  recurrent_kernel<<<NWG, 256, 0, stream>>>(
      encb, postwb, wihb, whhb, priorwb, postb, b_ih, b_hh, priorb, w_ih,
      actseq, gumbel, tau, hcur, hfp, decin, postlg, priolg, gi, gh,
      out_h, out_samp, sync);

  // KL loss
  kl_kernel<<<512, 256, 0, stream>>>(postlg, priolg, out_kl);

  // decoder + reward
  gemm_bf16_kernel<<<256, 256, 0, stream>>>(decin, 2048, decw1b, 2048, decb1,
                                            nullptr, dbuf, 4096, 1024, 2048, 1);
  gemm_bf16_kernel<<<256, 256, 0, stream>>>(dbuf, 1024, decw2b, 1024, decb2,
                                            out_recon, nullptr, 4096, 1024, 1024, 0);
  reward_kernel<<<512, 256, 0, stream>>>(decin, reww, rewb, out_rew);
}
